// KoLeoLoss_6511170421442
// MI455X (gfx1250) — matrix operand order repaired and checked by reading, not compiled, not run on hardware
//
#include <hip/hip_runtime.h>
#include <hip/hip_bf16.h>

typedef __attribute__((ext_vector_type(16))) __bf16 v16bf;
typedef __attribute__((ext_vector_type(8)))  __bf16 v8bf;
typedef __attribute__((ext_vector_type(8)))  float  v8f;
typedef __attribute__((ext_vector_type(4)))  int    v4i;

#define AS1 __attribute__((address_space(1)))
#define AS3 __attribute__((address_space(3)))

#define BN    8192      // rows
#define DK    512       // feature dim
#define BT    128       // block tile
#define KC    32        // K chunk (one bf16 WMMA depth)
#define NTB   (BN / BT) // 64 tile-blocks per side
#define LSTR  40        // LDS row stride in bf16 elements (32 data + 8 pad = 80B, conflict-free)
#define NCH   (DK / KC) // 16 K-chunks

#if defined(__has_builtin)
#if __has_builtin(__builtin_amdgcn_global_load_async_to_lds_b128)
#define HAVE_ASYNC_LDS 1
#endif
#endif
#ifndef HAVE_ASYNC_LDS
#define HAVE_ASYNC_LDS 0
#endif

union U8bf  { v8bf v;  __bf16 e[8]; };
union U16bf { v16bf v; v8bf h[2]; };

// ---------------------------------------------------------------------------
// Kernel 1: one-shot prep. Per row: sq[i]=||x_i||^2, minbuf[i]=+inf, and
// fp32 -> bf16 hi/lo split written to workspace (one wave per row).
// ---------------------------------------------------------------------------
__global__ void koleo_prep(const float* __restrict__ X,
                           unsigned short* __restrict__ Xhi_,
                           unsigned short* __restrict__ Xlo_,
                           float* __restrict__ sq,
                           unsigned* __restrict__ minbuf) {
    __bf16* Xhi = (__bf16*)Xhi_;
    __bf16* Xlo = (__bf16*)Xlo_;
    const int wave = threadIdx.x >> 5;
    const int lane = threadIdx.x & 31;
    const int row  = blockIdx.x * 8 + wave;

    const float* xr = X + (size_t)row * DK + lane * 16;
    float v[16];
    *(float4*)(v + 0)  = ((const float4*)xr)[0];
    *(float4*)(v + 4)  = ((const float4*)xr)[1];
    *(float4*)(v + 8)  = ((const float4*)xr)[2];
    *(float4*)(v + 12) = ((const float4*)xr)[3];

    float s = 0.0f;
    U8bf h[2], l[2];
#pragma unroll
    for (int i = 0; i < 16; ++i) {
        float f = v[i];
        s += f * f;
        __bf16 hb = (__bf16)f;
        h[i >> 3].e[i & 7] = hb;
        l[i >> 3].e[i & 7] = (__bf16)(f - (float)hb);
    }
    __bf16* ph = Xhi + (size_t)row * DK + lane * 16;
    __bf16* pl = Xlo + (size_t)row * DK + lane * 16;
    *(v8bf*)ph       = h[0].v;  *(v8bf*)(ph + 8) = h[1].v;
    *(v8bf*)pl       = l[0].v;  *(v8bf*)(pl + 8) = l[1].v;

#pragma unroll
    for (int m = 16; m >= 1; m >>= 1)
        s += __shfl_xor(s, m, 32);
    if (lane == 0) {
        sq[row] = s;
        minbuf[row] = 0x7f800000u; // +inf bits
    }
}

// ---------------------------------------------------------------------------
// Kernel 2: fused Gram-tile + row/col min over upper-triangular 128x128 tiles.
// bf16x3 WMMA (hi*hi + hi*lo + lo*hi, f32 accumulate); double-buffered LDS
// filled by async global->LDS DMA overlapping the matrix pipe.
// ---------------------------------------------------------------------------
__global__ __launch_bounds__(256, 2)
void koleo_gemm_min(const unsigned short* __restrict__ Xhi_,
                    const unsigned short* __restrict__ Xlo_,
                    const float* __restrict__ sq,
                    unsigned* __restrict__ minbuf) {
    const __bf16* Xhi = (const __bf16*)Xhi_;
    const __bf16* Xlo = (const __bf16*)Xlo_;

    // smem[buf][arr][BT*LSTR]; arr: 0=Ahi 1=Alo 2=Bhi 3=Blo
    __shared__ __bf16 smem[2 * 4 * BT * LSTR];
    __shared__ float  sSqI[BT];
    __shared__ float  sSqJ[BT];

    // map linear block id -> upper-triangular (bi, bj)
    int bi = 0, rem = (int)blockIdx.x, span = NTB;
    while (rem >= span) { rem -= span; ++bi; --span; }
    const int bj    = bi + rem;
    const int Ibase = bi * BT;
    const int Jbase = bj * BT;

    const int t     = threadIdx.x;
    const int wave  = t >> 5;
    const int lane  = t & 31;
    const int l15   = lane & 15;
    const int half  = lane >> 4;
    const int waveM = wave >> 1;  // 0..3 -> M offset waveM*32
    const int waveN = wave & 1;   // 0..1 -> N offset waveN*64

    if (t < BT) sSqI[t]      = sq[Ibase + t];
    else        sSqJ[t - BT] = sq[Jbase + (t - BT)];

    // staging: 512 16B-chunks per panel, 2 per thread
    auto stage = [&](int buf, int kcc) {
        const int kb = kcc * KC;
        const __bf16* srcs[4] = { Xhi + (size_t)Ibase * DK, Xlo + (size_t)Ibase * DK,
                                  Xhi + (size_t)Jbase * DK, Xlo + (size_t)Jbase * DK };
#pragma unroll
        for (int a = 0; a < 4; ++a) {
#pragma unroll
            for (int c = 0; c < 2; ++c) {
                const int ch  = t + c * 256;
                const int row = ch >> 2;
                const int seg = ch & 3;
                const __bf16* g = srcs[a] + (size_t)row * DK + kb + seg * 8;
                __bf16* l = &smem[((buf * 4 + a) * BT + row) * LSTR + seg * 8];
#if HAVE_ASYNC_LDS
                __builtin_amdgcn_global_load_async_to_lds_b128(
                    (AS1 v4i*)(void*)g, (AS3 v4i*)(void*)l, 0, 0);
#else
                *(v8bf*)l = *(const v8bf*)g;
#endif
            }
        }
    };

    v8f acc[8] = {}; // acc[nt*2 + mt]

    stage(0, 0);
#if HAVE_ASYNC_LDS
    asm volatile("s_wait_asynccnt 0x0" ::: "memory");
#endif
    __syncthreads();

    for (int kc = 0; kc < NCH; ++kc) {
        const int cur = kc & 1;
        if (kc + 1 < NCH) stage(cur ^ 1, kc + 1);

        const __bf16* pAhi = &smem[(cur * 4 + 0) * BT * LSTR];
        const __bf16* pAlo = &smem[(cur * 4 + 1) * BT * LSTR];
        const __bf16* pBhi = &smem[(cur * 4 + 2) * BT * LSTR];
        const __bf16* pBlo = &smem[(cur * 4 + 3) * BT * LSTR];

        // A (16x32): lanes 0-15 row=l15 K{0..7,16..23}; lanes 16-31 K{8..15,24..31}
        U16bf Ah[2], Al[2];
#pragma unroll
        for (int mt = 0; mt < 2; ++mt) {
            const int arow = (waveM * 32 + mt * 16 + l15) * LSTR;
            Ah[mt].h[0] = *(const v8bf*)(pAhi + arow + half * 8);
            Ah[mt].h[1] = *(const v8bf*)(pAhi + arow + 16 + half * 8);
            Al[mt].h[0] = *(const v8bf*)(pAlo + arow + half * 8);
            Al[mt].h[1] = *(const v8bf*)(pAlo + arow + 16 + half * 8);
        }
        // B (32x16, KxN): lane = N; lanes 0-15 K{0..7,16..23}; lanes 16-31 K{8..15,24..31}
        U16bf Bh[4], Bl[4];
#pragma unroll
        for (int nt = 0; nt < 4; ++nt) {
            const int brow = (waveN * 64 + nt * 16 + l15) * LSTR;
            Bh[nt].h[0] = *(const v8bf*)(pBhi + brow + half * 8);
            Bh[nt].h[1] = *(const v8bf*)(pBhi + brow + 16 + half * 8);
            Bl[nt].h[0] = *(const v8bf*)(pBlo + brow + half * 8);
            Bl[nt].h[1] = *(const v8bf*)(pBlo + brow + 16 + half * 8);
        }

#pragma unroll
        for (int nt = 0; nt < 4; ++nt) {
#pragma unroll
            for (int mt = 0; mt < 2; ++mt) {
                v8f c = acc[nt * 2 + mt];
                c = __builtin_amdgcn_wmma_f32_16x16x32_bf16(
                        false, Ah[mt].v, false, Bh[nt].v, (short)0, c, false, false);
                c = __builtin_amdgcn_wmma_f32_16x16x32_bf16(
                        false, Ah[mt].v, false, Bl[nt].v, (short)0, c, false, false);
                c = __builtin_amdgcn_wmma_f32_16x16x32_bf16(
                        false, Al[mt].v, false, Bh[nt].v, (short)0, c, false, false);
                acc[nt * 2 + mt] = c;
            }
        }

#if HAVE_ASYNC_LDS
        asm volatile("s_wait_asynccnt 0x0" ::: "memory");
#endif
        __syncthreads();
    }

    // ---- epilogue: d2 = sq_i + sq_j - 2*gram, mask diag, row & col mins ----
    const float FINF = __builtin_inff();
    float rowacc[2][8];
#pragma unroll
    for (int mt = 0; mt < 2; ++mt)
#pragma unroll
        for (int r = 0; r < 8; ++r) rowacc[mt][r] = FINF;
    float colmin[4] = {FINF, FINF, FINF, FINF};

#pragma unroll
    for (int nt = 0; nt < 4; ++nt) {
        const int jl  = waveN * 64 + nt * 16 + l15;
        const int jg  = Jbase + jl;
        const float sqj = sSqJ[jl];
#pragma unroll
        for (int mt = 0; mt < 2; ++mt) {
            v8f a = acc[nt * 2 + mt];
#pragma unroll
            for (int r = 0; r < 8; ++r) {
                const int m  = waveM * 32 + mt * 16 + half * 8 + r;
                const int ig = Ibase + m;
                float d2 = sSqI[m] + sqj - 2.0f * a[r];
                d2 = (ig == jg) ? FINF : fmaxf(d2, 0.0f);
                rowacc[mt][r] = fminf(rowacc[mt][r], d2);
                colmin[nt]    = fminf(colmin[nt], d2);
            }
        }
    }
    // row mins: reduce across the 16 lanes of each half; lanes 0/16 commit
#pragma unroll
    for (int mt = 0; mt < 2; ++mt) {
#pragma unroll
        for (int r = 0; r < 8; ++r) {
            float v = rowacc[mt][r];
            v = fminf(v, __shfl_xor(v, 1, 32));
            v = fminf(v, __shfl_xor(v, 2, 32));
            v = fminf(v, __shfl_xor(v, 4, 32));
            v = fminf(v, __shfl_xor(v, 8, 32));
            if (l15 == 0) {
                const int ig = Ibase + waveM * 32 + mt * 16 + half * 8 + r;
                atomicMin(&minbuf[ig], __float_as_uint(v));
            }
        }
    }
    // col mins (symmetry): combine halves; lanes 0-15 commit
#pragma unroll
    for (int nt = 0; nt < 4; ++nt) {
        float v = fminf(colmin[nt], __shfl_xor(colmin[nt], 16, 32));
        if (half == 0) {
            const int jg = Jbase + waveN * 64 + nt * 16 + l15;
            atomicMin(&minbuf[jg], __float_as_uint(v));
        }
    }
}

// ---------------------------------------------------------------------------
// Kernel 3: loss = mean(-log(sqrt(d2min) + eps))
// ---------------------------------------------------------------------------
__global__ void koleo_finalize(const unsigned* __restrict__ minbuf,
                               float* __restrict__ out) {
    __shared__ float red[256];
    float s = 0.0f;
    for (int i = threadIdx.x; i < BN; i += 256) {
        float nn = sqrtf(__uint_as_float(minbuf[i]));
        s += -logf(nn + 1e-8f);
    }
    red[threadIdx.x] = s;
    __syncthreads();
    for (int off = 128; off > 0; off >>= 1) {
        if (threadIdx.x < off) red[threadIdx.x] += red[threadIdx.x + off];
        __syncthreads();
    }
    if (threadIdx.x == 0) out[0] = red[0] * (1.0f / (float)BN);
}

extern "C" void kernel_launch(void* const* d_in, const int* in_sizes, int n_in,
                              void* d_out, int out_size, void* d_ws, size_t ws_size,
                              hipStream_t stream) {
    (void)in_sizes; (void)n_in; (void)out_size; (void)ws_size;
    const float* X = (const float*)d_in[0];

    // workspace layout: sq (32KB) | minbuf (32KB) | Xhi (8MB) | Xlo (8MB)
    float*          sq     = (float*)d_ws;
    unsigned*       minbuf = (unsigned*)((char*)d_ws + (32u << 10));
    unsigned short* Xhi    = (unsigned short*)((char*)d_ws + (64u << 10));
    unsigned short* Xlo    = (unsigned short*)((char*)d_ws + (64u << 10) + (size_t)BN * DK * 2);

    koleo_prep<<<BN / 8, 256, 0, stream>>>(X, Xhi, Xlo, sq, minbuf);
    const int nblocks = NTB * (NTB + 1) / 2; // 2080 upper-triangular tiles
    koleo_gemm_min<<<nblocks, 256, 0, stream>>>(Xhi, Xlo, sq, minbuf);
    koleo_finalize<<<1, 256, 0, stream>>>(minbuf, (float*)d_out);
}
